// ControllerNetwork_32701880992457
// MI455X (gfx1250) — matrix operand order repaired and checked by reading, not compiled, not run on hardware
//
#include <hip/hip_runtime.h>
#include <hip/hip_bf16.h>
#include <math.h>

#define H 4096
#define IN_DIM 96
#define NSTEPS 128
#define NUM_TYPES 32
#define NUM_LAYERS 64

typedef __attribute__((ext_vector_type(16))) __bf16 v16bf;
typedef __attribute__((ext_vector_type(8)))  __bf16 v8bf;
typedef __attribute__((ext_vector_type(8)))  float  v8f;
typedef __attribute__((ext_vector_type(8)))  unsigned short us8;

static __device__ __forceinline__ v16bf cat16(v8bf lo, v8bf hi) {
  return __builtin_shufflevector(lo, hi, 0,1,2,3,4,5,6,7,8,9,10,11,12,13,14,15);
}

// round-to-nearest-even fp32 -> bf16 (as ushort bit pattern)
static __device__ __forceinline__ unsigned short f2bf(float f) {
  unsigned int u = __float_as_uint(f);
  u += 0x7FFFu + ((u >> 16) & 1u);
  return (unsigned short)(u >> 16);
}

static __device__ __forceinline__ float sigm(float x) {
  return 1.0f / (1.0f + expf(-x));
}

// ---- one-time fp32 -> bf16 conversion of w_hh (128MB fits in 192MB L2) ----
__global__ __launch_bounds__(256) void convert_whh(const float* __restrict__ w,
                                                   unsigned short* __restrict__ o,
                                                   long long n) {
  long long i = ((long long)blockIdx.x * blockDim.x + threadIdx.x) * 4;
  if (i + 3 < n) {
    float4 f = *(const float4*)(w + i);
    ushort4 p;
    p.x = f2bf(f.x); p.y = f2bf(f.y); p.z = f2bf(f.z); p.w = f2bf(f.w);
    *(ushort4*)(o + i) = p;
  }
}

// ---- zero initial state (harness poisons ws) ----
__global__ __launch_bounds__(256) void init_state(float* c, float* h,
                                                  unsigned short* hbf0, float* x,
                                                  int* counter) {
  int i = blockIdx.x * blockDim.x + threadIdx.x;
  if (i < H) { c[i] = 0.0f; h[i] = 0.0f; hbf0[i] = 0; }
  if (i < IN_DIM) x[i] = 0.0f;
  if (i == 0 && blockIdx.x == 0) *counter = 0;
}

// ---- per-step: gates = W_ih x + W_hh h + biases; LSTM pointwise update ----
// grid = 256 blocks, block = 256 threads (8 waves); block owns 16 h-rows.
// wave -> (gate = w&3, K-half = w>>2)
__global__ __launch_bounds__(256) void gates_kernel(
    const float* __restrict__ w_ih,
    const float* __restrict__ w_hh,
    const unsigned short* __restrict__ wbf,   // bf16 copy of w_hh (may be null)
    const float* __restrict__ b_ih,
    const float* __restrict__ b_hh,
    const float* __restrict__ xbuf,           // 96 fp32
    const unsigned short* __restrict__ hbf_in,// 4096 bf16 (prev h)
    float* __restrict__ cbuf,
    float* __restrict__ hbuf,
    unsigned short* __restrict__ hbf_out) {
  __shared__ float s_y[2][4][16];  // [khalf][gate][hrow in block]
  __shared__ float s_g[4][16];
  __shared__ float s_x[IN_DIM];

  const int t = threadIdx.x;
  if (t < IN_DIM) s_x[t] = xbuf[t];

  const int lane = t & 31;
  const int wave = t >> 5;          // 0..7
  const int gate = wave & 3;
  const int khalf = wave >> 2;      // 0/1
  const int hrow_base = blockIdx.x * 16;
  const int grow_base = gate * H + hrow_base;
  const long long row = grow_base + (lane & 15);
  const int aoff = (lane < 16) ? 0 : 8;   // 16-bit A layout K split
  const int boff = (lane < 16) ? 0 : 8;   // 16-bit B layout K split (same as A)
  const int k0 = khalf * (H / 2);
  const int k1 = k0 + (H / 2);

  v8f acc = {};
  if (wbf) {
    const unsigned short* wr = wbf + row * H;
    for (int kb = k0; kb < k1; kb += 32) {
      v8bf alo = __builtin_bit_cast(v8bf, *(const us8*)(wr + kb + aoff));
      v8bf ahi = __builtin_bit_cast(v8bf, *(const us8*)(wr + kb + aoff + 16));
      v8bf blo = __builtin_bit_cast(v8bf, *(const us8*)(hbf_in + kb + boff));
      v8bf bhi = __builtin_bit_cast(v8bf, *(const us8*)(hbf_in + kb + boff + 16));
      acc = __builtin_amdgcn_wmma_f32_16x16x32_bf16(
          false, cat16(alo, ahi), false, cat16(blo, bhi), (short)0, acc, false, false);
    }
  } else {
    // fallback: stream fp32 weights, convert to bf16 in-register
    const float* wr = w_hh + row * H;
    for (int kb = k0; kb < k1; kb += 32) {
      v16bf a;
#pragma unroll
      for (int j = 0; j < 8; ++j) a[j] = (__bf16)wr[kb + aoff + j];
#pragma unroll
      for (int j = 0; j < 8; ++j) a[8 + j] = (__bf16)wr[kb + aoff + 16 + j];
      v8bf blo = __builtin_bit_cast(v8bf, *(const us8*)(hbf_in + kb + boff));
      v8bf bhi = __builtin_bit_cast(v8bf, *(const us8*)(hbf_in + kb + boff + 16));
      acc = __builtin_amdgcn_wmma_f32_16x16x32_bf16(
          false, a, false, cat16(blo, bhi), (short)0, acc, false, false);
    }
  }

  // B columns are replicated, so every column of D == matvec result.
  // C/D layout: VGPR r, lanes 0-15 -> M=r; lanes 16-31 -> M=r+8.
  if (lane == 0) {
#pragma unroll
    for (int r = 0; r < 8; ++r) s_y[khalf][gate][r] = acc[r];
  } else if (lane == 16) {
#pragma unroll
    for (int r = 0; r < 8; ++r) s_y[khalf][gate][8 + r] = acc[r];
  }
  __syncthreads();

  if (t < 64) {  // 4 gates x 16 rows: add K-halves, biases, w_ih @ x
    const int g2 = t >> 4, r = t & 15;
    const long long grow = (long long)g2 * H + hrow_base + r;
    const float* wi = w_ih + grow * IN_DIM;
    float v = s_y[0][g2][r] + s_y[1][g2][r] + b_ih[grow] + b_hh[grow];
    for (int j = 0; j < IN_DIM; ++j) v += wi[j] * s_x[j];
    s_g[g2][r] = v;
  }
  __syncthreads();

  if (t < 16) {  // LSTM pointwise
    const int hr = hrow_base + t;
    float ig = sigm(s_g[0][t]);
    float fg = sigm(s_g[1][t]);
    float gg = tanhf(s_g[2][t]);
    float og = sigm(s_g[3][t]);
    float c = fg * cbuf[hr] + ig * gg;
    cbuf[hr] = c;
    float h = og * tanhf(c);
    hbuf[hr] = h;
    hbf_out[hr] = f2bf(h);
  }
}

// ---- per-step: fc heads (96 parallel row blocks), last block finishes ----
// grid = 96 blocks x 256 threads. Block b computes row b of [fc1; fc2].
// Last-arriving block does softmax + -log1p(k*exp(-z)) + next-x + outputs.
__global__ __launch_bounds__(256) void fc_kernel(
    const float* __restrict__ fc1_w, const float* __restrict__ fc1_b,
    const float* __restrict__ fc2_w, const float* __restrict__ fc2_b,
    const float* __restrict__ h,
    float* __restrict__ ubuf,      // 96 fp32 scratch (pre-activation)
    int* __restrict__ counter,     // cross-block arrival counter (starts 0)
    float* __restrict__ xbuf, float* __restrict__ out1, float* __restrict__ out2,
    float kf) {
  __shared__ float s_p[8];
  __shared__ int s_last;
  const int t = threadIdx.x, lane = t & 31, wave = t >> 5;
  const int b = blockIdx.x;  // row 0..95

  const float* wr = (b < NUM_TYPES) ? (fc1_w + (long long)b * H)
                                    : (fc2_w + (long long)(b - NUM_TYPES) * H);
  float acc = 0.0f;
  for (int k = t; k < H; k += 256) acc += wr[k] * h[k];
#pragma unroll
  for (int off = 16; off > 0; off >>= 1) acc += __shfl_down(acc, off);
  if (lane == 0) s_p[wave] = acc;
  __syncthreads();
  if (t == 0) {
    float s = 0.0f;
#pragma unroll
    for (int i = 0; i < 8; ++i) s += s_p[i];
    float bias = (b < NUM_TYPES) ? fc1_b[b] : fc2_b[b - NUM_TYPES];
    ubuf[b] = s + bias;
    __threadfence();                       // publish ubuf[b]
    int old = atomicAdd(counter, 1);
    s_last = (old == IN_DIM - 1) ? 1 : 0;  // 96th arrival finishes
  }
  __syncthreads();
  if (!s_last) return;
  __threadfence();  // acquire all ubuf writes

  volatile const float* u = (volatile const float*)ubuf;
  if (t < NUM_TYPES) {  // one full wave: softmax over 32
    float v = u[t];
    float mx = v;
#pragma unroll
    for (int off = 16; off > 0; off >>= 1) mx = fmaxf(mx, __shfl_xor(mx, off));
    float e = expf(v - mx);
    float s = e;
#pragma unroll
    for (int off = 16; off > 0; off >>= 1) s += __shfl_xor(s, off);
    float r1 = e / s;
    out1[t] = r1;
    xbuf[t] = r1;
  } else if (t >= 32 && t < IN_DIM) {
    float z = u[t];
    float v = -log1pf(kf * expf(-z));
    out2[t - 32] = v;
    xbuf[t] = v;
  }
  __syncthreads();
  if (t == 0) *counter = 0;  // reset for next step's launch
}

extern "C" void kernel_launch(void* const* d_in, const int* in_sizes, int n_in,
                              void* d_out, int out_size, void* d_ws, size_t ws_size,
                              hipStream_t stream) {
  (void)in_sizes; (void)n_in; (void)out_size;
  const float* w_ih  = (const float*)d_in[0];
  const float* w_hh  = (const float*)d_in[1];
  const float* b_ih  = (const float*)d_in[2];
  const float* b_hh  = (const float*)d_in[3];
  const float* fc1_w = (const float*)d_in[4];
  const float* fc1_b = (const float*)d_in[5];
  const float* fc2_w = (const float*)d_in[6];
  const float* fc2_b = (const float*)d_in[7];
  float* out = (float*)d_out;

  char* ws = (char*)d_ws;
  unsigned short* hbf = (unsigned short*)ws;        // 2 x 4096 bf16 (double buffer)
  float* cbuf         = (float*)(ws + 32 * 1024);   // 4096 fp32
  float* hbuf         = (float*)(ws + 48 * 1024);   // 4096 fp32
  float* xbuf         = (float*)(ws + 64 * 1024);   // 96 fp32
  float* ubuf         = (float*)(ws + 68 * 1024);   // 96 fp32
  int*   counter      = (int*)  (ws + 72 * 1024);   // 1 int
  const size_t WBF_BYTES = (size_t)4 * H * (size_t)H * 2;  // 128 MB
  unsigned short* wbf = nullptr;
  if (ws_size >= 128 * 1024 + WBF_BYTES)
    wbf = (unsigned short*)(ws + 128 * 1024);

  if (wbf) {
    long long n = (long long)4 * H * H;
    convert_whh<<<(unsigned)(n / 4 / 256), 256, 0, stream>>>(w_hh, wbf, n);
  }
  init_state<<<16, 256, 0, stream>>>(cbuf, hbuf, hbf, xbuf, counter);

  for (int s = 0; s < NSTEPS; ++s) {
    const unsigned short* hin = hbf + (s & 1) * H;
    unsigned short* hout = hbf + ((s + 1) & 1) * H;
    gates_kernel<<<H / 16, 256, 0, stream>>>(w_ih, w_hh, wbf, b_ih, b_hh, xbuf,
                                             hin, cbuf, hbuf, hout);
    fc_kernel<<<IN_DIM, 256, 0, stream>>>(fc1_w, fc1_b, fc2_w, fc2_b, hbuf,
                                          ubuf, counter, xbuf,
                                          out + (size_t)s * NUM_TYPES,
                                          out + (size_t)NSTEPS * NUM_TYPES + (size_t)s * NUM_LAYERS,
                                          (float)(s >> 1));
  }
}
